// BertLikeSelfAttention_893353197834
// MI455X (gfx1250) — compile-verified, matrix operand order checked by reading, not run on hardware
//
#include <hip/hip_runtime.h>
#include <math.h>

#define B_    4
#define S_    2048
#define HID_  1024
#define NH_   16
#define HD_   64
#define CAP_  30.0f
#define EPS_  1e-6f

typedef __bf16 bf16;
typedef __attribute__((ext_vector_type(16))) __bf16 v16bf;
typedef __attribute__((ext_vector_type(8)))  __bf16 bf16x8;
typedef __attribute__((ext_vector_type(8)))  float  v8f;
typedef __attribute__((ext_vector_type(4)))  float  f32x4;
typedef __attribute__((ext_vector_type(4)))  unsigned int u32x4;
typedef __attribute__((ext_vector_type(8)))  unsigned int u32x8;

// ---- CDNA5 async copy: global -> LDS, 16B per lane, tracked by ASYNCcnt ----
__device__ __forceinline__ void async_copy_b128(bf16* lds_ptr, const bf16* gptr) {
    // low 32 bits of a flat shared pointer == group-relative LDS offset (VDST operand)
    unsigned lds_off = (unsigned)(unsigned long long)lds_ptr;
    asm volatile("global_load_async_to_lds_b128 %0, %1, off"
                 :: "v"(lds_off), "v"(gptr) : "memory");
}

template <int N>
__device__ __forceinline__ void wait_asynccnt() {
    asm volatile("s_wait_asynccnt %0" :: "n"(N) : "memory");
}

// ---- CDNA5 Tensor Data Mover: descriptor-driven 2D tile load to LDS -------
// Builds a D# per ISA 8.3/8.4 (count=1, type=2, data_size=2B, no pad/iterate,
// groups 2/3 zero for a 2D tile) and issues TENSOR_LOAD_TO_LDS. All operands
// are block-uniform -> SGPR tuples. Tracked by TENSORcnt.
__device__ __forceinline__ void tdm_load_2d(const bf16* lds_ptr, const bf16* gptr,
                                            unsigned tile_w, unsigned tile_h,
                                            unsigned tensor_w, unsigned tensor_h,
                                            unsigned row_stride) {
    unsigned long long ga = (unsigned long long)gptr;
    u32x4 g0;
    g0[0] = 1u;                                               // count=1, user descriptor
    g0[1] = (unsigned)(unsigned long long)lds_ptr;            // lds_addr (bytes)
    g0[2] = (unsigned)ga;                                     // global_addr[31:0]
    g0[3] = (unsigned)((ga >> 32) & 0x1FFFFFFu) | (2u << 30); // global_addr[56:32] | type=2
    u32x8 g1;
    g1[0] = (1u << 16);                                       // data_size=2B; mask/pad/iterate=0
    g1[1] = (tensor_w & 0xFFFFu) << 16;                       // tensor_dim0[15:0]
    g1[2] = ((tensor_w >> 16) & 0xFFFFu) | ((tensor_h & 0xFFFFu) << 16);
    g1[3] = ((tensor_h >> 16) & 0xFFFFu) | ((tile_w & 0xFFFFu) << 16);
    g1[4] = (tile_h & 0xFFFFu);                               // tile_dim1; tile_dim2=0
    g1[5] = row_stride;                                       // tensor_dim0_stride[31:0]
    g1[6] = 0u;                                               // stride hi + dim1_stride lo
    g1[7] = 0u;
    u32x4 gz = {0u, 0u, 0u, 0u};
    asm volatile("tensor_load_to_lds %0, %1, %2, %3"
                 :: "s"(g0), "s"(g1), "s"(gz), "s"(gz) : "memory");
}

// Load a 16-element bf16 WMMA fragment: lane's low 8 k-values at p, high 8 at p+16
// (caller folds the lane<16 vs lane>=16 offset of +8 into p). ISA 7.12.2 layout.
__device__ __forceinline__ v16bf load_frag(const bf16* p) {
    bf16x8 lo = *(const bf16x8*)(p);
    bf16x8 hi = *(const bf16x8*)(p + 16);
    v16bf r;
#pragma unroll
    for (int i = 0; i < 8; ++i) { r[i] = lo[i]; r[i + 8] = hi[i]; }
    return r;
}

__device__ __forceinline__ v8f wmma_bf16(v16bf a, v16bf b, v8f c) {
    return __builtin_amdgcn_wmma_f32_16x16x32_bf16(
        /*neg_a=*/false, a, /*neg_b=*/false, b,
        /*c_mod=*/(short)0, c, /*reuse_a=*/false, /*reuse_b=*/false);
}

// ---------------- fp32 -> bf16 conversion (4-wide) ----------------
__global__ __launch_bounds__(256)
void cvt_f32_bf16(const float* __restrict__ src, bf16* __restrict__ dst, int n) {
    int i = (blockIdx.x * 256 + threadIdx.x) * 4;
    if (i + 3 < n) {
        f32x4 v = *(const f32x4*)(src + i);
        dst[i + 0] = (bf16)v.x;
        dst[i + 1] = (bf16)v.y;
        dst[i + 2] = (bf16)v.z;
        dst[i + 3] = (bf16)v.w;
    }
}

// ---------------- QKV projection GEMM -----------------------------
// O[row, col] = sum_h X[row,h] * W[col,h] + bias[col]   (torch Linear)
// grid = (M/64, N/128, 3); block = 256 (8 waves in 4x2); wave -> 16x64 tile.
// X tile (64x32) and W tile (128x32) async-staged into double-buffered LDS.
// Q,K stored [B,H,S,D] bf16; V stored transposed [B,H,D,S] bf16.
__global__ __launch_bounds__(256)
void qkv_gemm(const bf16* __restrict__ X,
              const bf16* __restrict__ Wqb, const bf16* __restrict__ Wkb,
              const bf16* __restrict__ Wvb,
              const float* __restrict__ bq, const float* __restrict__ bk,
              const float* __restrict__ bv,
              bf16* __restrict__ Oq, bf16* __restrict__ Ok, bf16* __restrict__ Ov) {
    __shared__ __align__(16) bf16 xbuf[2][64 * 32];    // 2 x 4KB
    __shared__ __align__(16) bf16 wbuf[2][128 * 32];   // 2 x 8KB

    const int tid  = threadIdx.x;
    const int lane = tid & 31;
    const int wave = tid >> 5;
    const int z    = blockIdx.z;

    const bf16*  W    = (z == 0) ? Wqb : (z == 1) ? Wkb : Wvb;
    const float* bias = (z == 0) ? bq  : (z == 1) ? bk  : bv;
    bf16*        Out  = (z == 0) ? Oq  : (z == 1) ? Ok  : Ov;

    const int wave_row = wave >> 1;                    // 0..3
    const int wave_col = wave & 1;                     // 0..1
    const int rowblk   = blockIdx.x * 64;
    const int colblk   = blockIdx.y * 128;
    const int row0     = rowblk + wave_row * 16;
    const int loff     = (lane & 16) ? 8 : 0;
    const int lrow     = lane & 15;

    // per-thread staging coordinates (3 async b128 per wave per k-step)
    const int xe = tid * 8;                            // 64x32 tile, 8 bf16/thread
    const int xr = xe >> 5, xc = xe & 31;
    const int wr = tid >> 1, wc = (tid & 1) * 16;      // 128x32 tile, 16 bf16/thread

    // preload k0 = 0 into buffer 0
    async_copy_b128(&xbuf[0][xe],               &X[(rowblk + xr) * HID_ + xc]);
    async_copy_b128(&wbuf[0][wr * 32 + wc],     &W[(colblk + wr) * HID_ + wc]);
    async_copy_b128(&wbuf[0][wr * 32 + wc + 8], &W[(colblk + wr) * HID_ + wc + 8]);

    v8f acc[4] = {};
    int cur = 0;

    for (int k0 = 0; k0 < HID_; k0 += 32) {
        if (k0 + 32 < HID_) {
            const int kn = k0 + 32;
            async_copy_b128(&xbuf[cur ^ 1][xe],               &X[(rowblk + xr) * HID_ + kn + xc]);
            async_copy_b128(&wbuf[cur ^ 1][wr * 32 + wc],     &W[(colblk + wr) * HID_ + kn + wc]);
            async_copy_b128(&wbuf[cur ^ 1][wr * 32 + wc + 8], &W[(colblk + wr) * HID_ + kn + wc + 8]);
            wait_asynccnt<3>();                        // current tile arrived; next in flight
        } else {
            wait_asynccnt<0>();
        }
        __syncthreads();

        // batch all fragment loads, then chain WMMAs (single DS wait)
        v16bf afrag = load_frag(&xbuf[cur][(wave_row * 16 + lrow) * 32 + loff]);
        v16bf bfr[4];
#pragma unroll
        for (int nt = 0; nt < 4; ++nt)
            bfr[nt] = load_frag(&wbuf[cur][(wave_col * 64 + nt * 16 + lrow) * 32 + loff]);
#pragma unroll
        for (int nt = 0; nt < 4; ++nt)
            acc[nt] = wmma_bf16(afrag, bfr[nt], acc[nt]);

        __syncthreads();                               // readers done before buffer reuse
        cur ^= 1;
    }

    // epilogue: bias add, cvt to bf16, scatter into head-major layouts
#pragma unroll
    for (int nt = 0; nt < 4; ++nt) {
        int col = colblk + wave_col * 64 + nt * 16 + lrow;
        float bsv = bias[col];
        int h = col >> 6, d = col & 63;
#pragma unroll
        for (int r = 0; r < 8; ++r) {
            int row = row0 + r + ((lane >> 4) << 3);
            int b = row >> 11, s = row & (S_ - 1);
            float v = acc[nt][r] + bsv;
            long idx;
            if (z == 2) idx = ((long)(b * NH_ + h) * HD_ + d) * S_ + s;   // V transposed
            else        idx = ((long)(b * NH_ + h) * S_ + s) * HD_ + d;   // Q, K
            Out[idx] = (bf16)v;
        }
    }
}

// ---------------- fused attention ---------------------------------
// grid = (S/128, H, B); block = 256 (8 waves); wave -> 16 q-rows.
// 32-key chunks staged by the Tensor Data Mover (one wave issues the
// descriptor-driven DMA, TENSORcnt + barrier synchronize); scores ->
// cap/relu -> PV accumulate; denominator via WMMA against all-ones B.
__global__ __launch_bounds__(256)
void attn_kernel(const bf16* __restrict__ Q, const bf16* __restrict__ K,
                 const bf16* __restrict__ Vt, const float* __restrict__ mask,
                 float* __restrict__ out) {
    __shared__ __align__(16) bf16 kbuf[2][32 * 64];    // K chunk  [kk][d], 2 x 4KB
    __shared__ __align__(16) bf16 vbuf[2][64 * 32];    // V chunk  [d][kk], 2 x 4KB
    __shared__ __align__(16) bf16 sbuf[8][16 * 32];    // per-wave relu-score staging, 8KB

    const int tid  = threadIdx.x;
    const int lane = tid & 31;
    const int wave = tid >> 5;
    const int b    = blockIdx.z;
    const int h    = blockIdx.y;
    const int bh   = b * NH_ + h;
    const int q0   = blockIdx.x * 128 + wave * 16;
    const int loff = (lane & 16) ? 8 : 0;
    const int lrow = lane & 15;

    const bf16*  kg   = K  + (long)bh * S_ * HD_;
    const bf16*  vg   = Vt + (long)bh * HD_ * S_;
    const float* mrow = mask + (long)b * S_;

    // Q fragments for this wave's 16 rows (D=64 -> two k-steps), kept in regs
    const bf16* qrow = Q + ((long)bh * S_ + q0 + lrow) * HD_;
    v16bf qf0 = load_frag(qrow + 0  + loff);
    v16bf qf1 = load_frag(qrow + 32 + loff);

    v16bf ones;
#pragma unroll
    for (int i = 0; i < 16; ++i) ones[i] = (bf16)1.0f;

    v8f cacc[4] = {};
    v8f dacc = {};

    // preload chunk 0 into buffer 0 via TDM (EXEC-independent, one wave issues)
    if (wave == 0) {
        tdm_load_2d(&kbuf[0][0], kg, /*tile*/ HD_, 32, /*tensor*/ HD_, 32, /*stride*/ HD_);
        tdm_load_2d(&vbuf[0][0], vg, /*tile*/ 32, HD_, /*tensor*/ 32, HD_, /*stride*/ S_);
    }

    int cur = 0;
    for (int j = 0; j < S_; j += 32) {
        if (wave == 0) {
            if (j + 32 < S_) {
                tdm_load_2d(&kbuf[cur ^ 1][0], kg + (long)(j + 32) * HD_,
                            HD_, 32, HD_, 32, HD_);
                tdm_load_2d(&vbuf[cur ^ 1][0], vg + (j + 32),
                            32, HD_, 32, HD_, S_);
                __builtin_amdgcn_s_wait_tensorcnt((short)2);  // current chunk arrived
            } else {
                __builtin_amdgcn_s_wait_tensorcnt((short)0);
            }
        }
        __syncthreads();

        // scores: 16 q-rows x 32 keys (two 16x16 C tiles, K-depth 64)
        v8f sc0 = {}, sc1 = {};
        {
            const bf16* kp0 = &kbuf[cur][(0  + lrow) * 64];
            const bf16* kp1 = &kbuf[cur][(16 + lrow) * 64];
            v16bf k00 = load_frag(kp0 + 0  + loff);
            v16bf k01 = load_frag(kp0 + 32 + loff);
            v16bf k10 = load_frag(kp1 + 0  + loff);
            v16bf k11 = load_frag(kp1 + 32 + loff);
            sc0 = wmma_bf16(qf0, k00, sc0);
            sc0 = wmma_bf16(qf1, k01, sc0);
            sc1 = wmma_bf16(qf0, k10, sc1);
            sc1 = wmma_bf16(qf1, k11, sc1);
        }

        // scale + mask + tanh soft-cap + relu; stage as bf16 A-matrix
#pragma unroll
        for (int nt = 0; nt < 2; ++nt) {
            v8f scv = nt ? sc1 : sc0;
            float mv = mrow[j + nt * 16 + lrow];
#pragma unroll
            for (int r = 0; r < 8; ++r) {
                float s = scv[r] * 0.125f + mv;         // /sqrt(64) + mask
                s = tanhf(s * (1.0f / CAP_)) * CAP_;    // soft cap
                s = s > 0.0f ? s : 0.0f;                // relu
                int m = r + ((lane >> 4) << 3);
                sbuf[wave][m * 32 + nt * 16 + lrow] = (bf16)s;
            }
        }
        __syncthreads();

        // relu-scores as A fragment; batch V fragments, then WMMA chain
        v16bf sf = load_frag(&sbuf[wave][lrow * 32 + loff]);
        v16bf vf[4];
#pragma unroll
        for (int dt = 0; dt < 4; ++dt)
            vf[dt] = load_frag(&vbuf[cur][(dt * 16 + lrow) * 32 + loff]);
        dacc = wmma_bf16(sf, ones, dacc);               // row-sum via ones-matrix
#pragma unroll
        for (int dt = 0; dt < 4; ++dt)
            cacc[dt] = wmma_bf16(sf, vf[dt], cacc[dt]);

        __syncthreads();                                // readers done before buffer reuse
        cur ^= 1;
    }

    // normalize and write context: out[b, s, h*64 + d]  (fp32)
#pragma unroll
    for (int r = 0; r < 8; ++r) {
        int m = r + ((lane >> 4) << 3);
        int s = q0 + m;
        float inv = 1.0f / (dacc[r] + EPS_);
#pragma unroll
        for (int dt = 0; dt < 4; ++dt) {
            out[((long)(b * S_ + s)) * HID_ + h * HD_ + dt * 16 + lrow] = cacc[dt][r] * inv;
        }
    }
}

// ---------------- host launcher -----------------------------------
extern "C" void kernel_launch(void* const* d_in, const int* in_sizes, int n_in,
                              void* d_out, int out_size, void* d_ws, size_t ws_size,
                              hipStream_t stream) {
    (void)in_sizes; (void)n_in; (void)out_size; (void)ws_size;

    const float* hs   = (const float*)d_in[0];
    const float* mask = (const float*)d_in[1];
    const float* Wq   = (const float*)d_in[2];
    const float* bq   = (const float*)d_in[3];
    const float* Wk   = (const float*)d_in[4];
    const float* bk   = (const float*)d_in[5];
    const float* Wv   = (const float*)d_in[6];
    const float* bv   = (const float*)d_in[7];
    float* out = (float*)d_out;

    char* ws = (char*)d_ws;
    bf16* hsb = (bf16*)ws; ws += (size_t)B_ * S_ * HID_ * sizeof(bf16);
    bf16* wqb = (bf16*)ws; ws += (size_t)HID_ * HID_ * sizeof(bf16);
    bf16* wkb = (bf16*)ws; ws += (size_t)HID_ * HID_ * sizeof(bf16);
    bf16* wvb = (bf16*)ws; ws += (size_t)HID_ * HID_ * sizeof(bf16);
    bf16* Qb  = (bf16*)ws; ws += (size_t)B_ * S_ * HID_ * sizeof(bf16);
    bf16* Kb  = (bf16*)ws; ws += (size_t)B_ * S_ * HID_ * sizeof(bf16);
    bf16* Vtb = (bf16*)ws; ws += (size_t)B_ * S_ * HID_ * sizeof(bf16);

    const int n_hs = B_ * S_ * HID_;
    const int n_w  = HID_ * HID_;
    cvt_f32_bf16<<<n_hs / 1024, 256, 0, stream>>>(hs, hsb, n_hs);
    cvt_f32_bf16<<<n_w  / 1024, 256, 0, stream>>>(Wq, wqb, n_w);
    cvt_f32_bf16<<<n_w  / 1024, 256, 0, stream>>>(Wk, wkb, n_w);
    cvt_f32_bf16<<<n_w  / 1024, 256, 0, stream>>>(Wv, wvb, n_w);

    dim3 gg((B_ * S_) / 64, HID_ / 128, 3);
    qkv_gemm<<<gg, 256, 0, stream>>>(hsb, wqb, wkb, wvb, bq, bk, bv, Qb, Kb, Vtb);

    dim3 ga(S_ / 128, NH_, B_);
    attn_kernel<<<ga, 256, 0, stream>>>(Qb, Kb, Vtb, mask, out);
}
